// NCM_8830452761219
// MI455X (gfx1250) — matrix operand order repaired and checked by reading, not compiled, not run on hardware
//
#include <hip/hip_runtime.h>

// ---------------------------------------------------------------------------
// Types for CDNA5 WMMA (wave32): bf16 16x16x32, f32 accumulate
// ---------------------------------------------------------------------------
typedef __bf16 bf16_t;
typedef __attribute__((ext_vector_type(8)))  bf16_t v8bf;
typedef __attribute__((ext_vector_type(16))) bf16_t v16bf;
typedef __attribute__((ext_vector_type(8)))  float  v8f;

#define BM 128
#define BN 128
#define BK 32
#define LDSTR 40          // bf16 elems per LDS row: 32 + 8 pad -> 80B stride (16B aligned, conflict-free)
#define LDSU  (LDSTR / 2) // row stride in uints

// ---------------------------------------------------------------------------
// f32 -> split bf16 (hi, lo). hi uses round-to-nearest-away kept in the HIGH
// half of the word: it is simultaneously the packed bf16 bits and the exact
// f32 bits of hi. Residual r = f - hi is EXACT (Sterbenz: hi in [f/2, 2f]),
// so the hi rounding rule costs nothing; lo absorbs it.
// Pair-packing of two high halves is one v_perm_b32.
// ---------------------------------------------------------------------------
static __device__ __forceinline__ unsigned pack_hi16(unsigned hiWordY, unsigned hiWordX) {
  // result = { Y[31:16], X[31:16] }
  return __builtin_amdgcn_perm(hiWordY, hiWordX, 0x07060302u);
}

static __device__ __forceinline__ void cvt_store4(unsigned* hiBuf, unsigned* loBuf,
                                                  int offU, float4 v) {
  unsigned h0 = (__float_as_uint(v.x) + 0x8000u) & 0xFFFF0000u;
  unsigned h1 = (__float_as_uint(v.y) + 0x8000u) & 0xFFFF0000u;
  unsigned h2 = (__float_as_uint(v.z) + 0x8000u) & 0xFFFF0000u;
  unsigned h3 = (__float_as_uint(v.w) + 0x8000u) & 0xFFFF0000u;
  float r0 = v.x - __uint_as_float(h0);
  float r1 = v.y - __uint_as_float(h1);
  float r2 = v.z - __uint_as_float(h2);
  float r3 = v.w - __uint_as_float(h3);
  unsigned g0 = __float_as_uint(r0) + 0x8000u;   // no mask needed: perm picks bytes 2-3
  unsigned g1 = __float_as_uint(r1) + 0x8000u;
  unsigned g2 = __float_as_uint(r2) + 0x8000u;
  unsigned g3 = __float_as_uint(r3) + 0x8000u;
  uint2 hp; hp.x = pack_hi16(h1, h0); hp.y = pack_hi16(h3, h2);
  uint2 lp; lp.x = pack_hi16(g1, g0); lp.y = pack_hi16(g3, g2);
  *(uint2*)(hiBuf + offU) = hp;
  *(uint2*)(loBuf + offU) = lp;
}

// A fragment (16x32 bf16): lane<16 -> M=lane, K chunks [kh,kh+8) and [kh+16,kh+24), kh=(lane>=16)*8
static __device__ __forceinline__ v16bf ldfragA(const unsigned* sU, int mrow, int khalf) {
  const unsigned short* s = (const unsigned short*)sU;
  v8bf a = *(const v8bf*)(s + mrow * LDSTR + khalf);
  v8bf b = *(const v8bf*)(s + mrow * LDSTR + khalf + 16);
  return __builtin_shufflevector(a, b, 0,1,2,3,4,5,6,7,8,9,10,11,12,13,14,15);
}
// B fragment (32x16 bf16): lane holds column N=lane%16, K chunks [kh,kh+8) and [kh+16,kh+24), kh=(lane>=16)*8
static __device__ __forceinline__ v16bf ldfragB(const unsigned* sU, int ncol, int kh) {
  const unsigned short* s = (const unsigned short*)sU;
  v8bf a = *(const v8bf*)(s + ncol * LDSTR + kh);
  v8bf b = *(const v8bf*)(s + ncol * LDSTR + kh + 16);
  return __builtin_shufflevector(a, b, 0,1,2,3,4,5,6,7,8,9,10,11,12,13,14,15);
}

// ---------------------------------------------------------------------------
// Pass 1a: rinv[n] = 1 / ||features[n,:] - mean||
// ---------------------------------------------------------------------------
__global__ __launch_bounds__(256) void rownorm_kernel(const float* __restrict__ f,
                                                      const float* __restrict__ mean,
                                                      float* __restrict__ rinv, int D) {
  const int n = blockIdx.x;
  const int t = threadIdx.x;
  float s = 0.f;
  for (int d = t * 4; d < D; d += 256 * 4) {
    float4 a = *(const float4*)(f + (size_t)n * D + d);
    float4 m = *(const float4*)(mean + d);
    float dx = a.x - m.x, dy = a.y - m.y, dz = a.z - m.z, dw = a.w - m.w;
    s += dx * dx + dy * dy + dz * dz + dw * dw;
  }
  __shared__ float red[256];
  red[t] = s; __syncthreads();
  for (int off = 128; off > 0; off >>= 1) {
    if (t < off) red[t] += red[t + off];
    __syncthreads();
  }
  if (t == 0) rinv[n] = 1.0f / sqrtf(red[0]);
}

// ---------------------------------------------------------------------------
// Pass 1b: mc[c] = mean . cent[c,:],  cc[c] = ||cent[c,:]||^2
// ---------------------------------------------------------------------------
__global__ __launch_bounds__(256) void centstats_kernel(const float* __restrict__ cent,
                                                        const float* __restrict__ mean,
                                                        float* __restrict__ mc,
                                                        float* __restrict__ cc, int D) {
  const int c = blockIdx.x;
  const int t = threadIdx.x;
  float sm = 0.f, sc = 0.f;
  for (int d = t * 4; d < D; d += 256 * 4) {
    float4 v = *(const float4*)(cent + (size_t)c * D + d);
    float4 m = *(const float4*)(mean + d);
    sm += v.x * m.x + v.y * m.y + v.z * m.z + v.w * m.w;
    sc += v.x * v.x + v.y * v.y + v.z * v.z + v.w * v.w;
  }
  __shared__ float redm[256];
  __shared__ float redc[256];
  redm[t] = sm; redc[t] = sc; __syncthreads();
  for (int off = 128; off > 0; off >>= 1) {
    if (t < off) { redm[t] += redm[t + off]; redc[t] += redc[t + off]; }
    __syncthreads();
  }
  if (t == 0) { mc[c] = redm[0]; cc[c] = redc[0]; }
}

// ---------------------------------------------------------------------------
// Main: split-bf16 WMMA GEMM (G = F . C^T) fused with L2-distance epilogue
//   out[n,c] = -sqrt(max(1 + cc[c] - 2*(G - mc[c])*rinv[n], 1e-12))
// Double-buffered LDS, one barrier per k-step.
// ---------------------------------------------------------------------------
__global__ __launch_bounds__(256) void l2dist_wmma_kernel(
    const float* __restrict__ Af, const float* __restrict__ Bf,
    const float* __restrict__ rinv, const float* __restrict__ mc,
    const float* __restrict__ cc, float* __restrict__ out,
    int N, int C, int D) {
  __shared__ __align__(16) unsigned sAhi[2][BM * LDSU];
  __shared__ __align__(16) unsigned sAlo[2][BM * LDSU];
  __shared__ __align__(16) unsigned sBhi[2][BN * LDSU];
  __shared__ __align__(16) unsigned sBlo[2][BN * LDSU];
  __shared__ float sRinv[BM];
  __shared__ float sMc[BN];
  __shared__ float sCc[BN];

  const int tid  = threadIdx.x;
  const int lane = tid & 31;
  const int wave = tid >> 5;
  const int wm   = wave & 3;   // 4 waves along M: 32 rows each
  const int wn   = wave >> 2;  // 2 waves along N: 64 cols each
  const int brow = blockIdx.y * BM;
  const int bcol = blockIdx.x * BN;

  if (tid < BM) sRinv[tid] = rinv[brow + tid];
  if (tid < BN) { sMc[tid] = mc[bcol + tid]; sCc[tid] = cc[bcol + tid]; }

  // Coalesced tile loads: 128 rows x 8 float4/row, 256 threads x 4 iters
  const int lrow = tid >> 3;         // 0..31
  const int lcol = (tid & 7) * 4;    // 0,4,...,28 (elements)
  const int offU0 = lrow * LDSU + (lcol >> 1);  // uint offset within a stage
  const float* gA = Af + (size_t)(brow + lrow) * D + lcol;
  const float* gB = Bf + (size_t)(bcol + lrow) * D + lcol;

  float4 ra[4], rb[4];
  const int NT = D / BK;

  v8f acc[2][4];
  const v8f vzero = {0.f, 0.f, 0.f, 0.f, 0.f, 0.f, 0.f, 0.f};
#pragma unroll
  for (int s = 0; s < 2; ++s)
#pragma unroll
    for (int t = 0; t < 4; ++t) acc[s][t] = vzero;

  // prologue: stage tile 0 into buffer 0
#pragma unroll
  for (int i = 0; i < 4; ++i) {
    ra[i] = *(const float4*)(gA + (size_t)(i * 32) * D);
    rb[i] = *(const float4*)(gB + (size_t)(i * 32) * D);
  }
#pragma unroll
  for (int i = 0; i < 4; ++i) {
    const int off = offU0 + (i * 32) * LDSU;
    cvt_store4(sAhi[0], sAlo[0], off, ra[i]);
    cvt_store4(sBhi[0], sBlo[0], off, rb[i]);
  }
  __syncthreads();

  int stage = 0;
  for (int kt = 0; kt < NT; ++kt) {
    const bool more = (kt + 1 < NT);
    // issue next tile's global loads early (latency hidden by WMMA below)
    if (more) {
      const size_t ko = (size_t)(kt + 1) * BK;
#pragma unroll
      for (int i = 0; i < 4; ++i) {
        ra[i] = *(const float4*)(gA + (size_t)(i * 32) * D + ko);
        rb[i] = *(const float4*)(gB + (size_t)(i * 32) * D + ko);
      }
    }

    // fragments + WMMA: 2 M-subtiles x 4 N-subtiles, 3 bf16 WMMAs each
    const unsigned* bufAhi = sAhi[stage];
    const unsigned* bufAlo = sAlo[stage];
    const unsigned* bufBhi = sBhi[stage];
    const unsigned* bufBlo = sBlo[stage];
    const int akh = (lane >> 4) * 8;
    const int bkh = (lane >> 4) * 8;
    v16bf aHi[2], aLo[2];
#pragma unroll
    for (int s = 0; s < 2; ++s) {
      const int mrow = wm * 32 + s * 16 + (lane & 15);
      aHi[s] = ldfragA(bufAhi, mrow, akh);
      aLo[s] = ldfragA(bufAlo, mrow, akh);
    }
#pragma unroll
    for (int t = 0; t < 4; ++t) {
      const int ncol = wn * 64 + t * 16 + (lane & 15);
      v16bf bHi = ldfragB(bufBhi, ncol, bkh);
      v16bf bLo = ldfragB(bufBlo, ncol, bkh);
#pragma unroll
      for (int s = 0; s < 2; ++s) {
        acc[s][t] = __builtin_amdgcn_wmma_f32_16x16x32_bf16(
            false, aHi[s], false, bHi, (short)0, acc[s][t], false, false);
        acc[s][t] = __builtin_amdgcn_wmma_f32_16x16x32_bf16(
            false, aHi[s], false, bLo, (short)0, acc[s][t], false, false);
        acc[s][t] = __builtin_amdgcn_wmma_f32_16x16x32_bf16(
            false, aLo[s], false, bHi, (short)0, acc[s][t], false, false);
      }
    }

    // convert + store next tile into the other buffer (free: its readers
    // finished before the barrier that ended the previous iteration)
    if (more) {
      const int ns = stage ^ 1;
#pragma unroll
      for (int i = 0; i < 4; ++i) {
        const int off = offU0 + (i * 32) * LDSU;
        cvt_store4(sAhi[ns], sAlo[ns], off, ra[i]);
        cvt_store4(sBhi[ns], sBlo[ns], off, rb[i]);
      }
    }
    __syncthreads();
    stage ^= 1;
  }

  // Epilogue: C/D layout => VGPR j, lane<16: M=j, lane>=16: M=j+8; N=lane%16
  const int mofs = (lane >> 4) << 3;
#pragma unroll
  for (int s = 0; s < 2; ++s) {
#pragma unroll
    for (int t = 0; t < 4; ++t) {
      const int ncol = wn * 64 + t * 16 + (lane & 15);
      const float mcv = sMc[ncol];
      const float ccv = sCc[ncol];
#pragma unroll
      for (int j = 0; j < 8; ++j) {
        const int mrow = wm * 32 + s * 16 + j + mofs;
        const float g  = acc[s][t][j];
        const float xc = (g - mcv) * sRinv[mrow];
        const float sq = fmaxf(1.0f + ccv - 2.0f * xc, 1e-12f);
        out[(size_t)(brow + mrow) * C + (bcol + ncol)] = -sqrtf(sq);
      }
    }
  }
}

// ---------------------------------------------------------------------------
// Launcher
// ---------------------------------------------------------------------------
extern "C" void kernel_launch(void* const* d_in, const int* in_sizes, int n_in,
                              void* d_out, int out_size, void* d_ws, size_t ws_size,
                              hipStream_t stream) {
  const float* features  = (const float*)d_in[0];
  const float* mean      = (const float*)d_in[1];
  const float* centroids = (const float*)d_in[2];
  float* out = (float*)d_out;

  const int D = in_sizes[1];
  const int N = in_sizes[0] / D;
  const int C = in_sizes[2] / D;

  float* rinv = (float*)d_ws;       // [N]
  float* mc   = rinv + N;           // [C]
  float* cc   = mc + C;             // [C]

  rownorm_kernel<<<N, 256, 0, stream>>>(features, mean, rinv, D);
  centstats_kernel<<<C, 256, 0, stream>>>(centroids, mean, mc, cc, D);

  dim3 grid(C / BN, N / BM);
  l2dist_wmma_kernel<<<grid, 256, 0, stream>>>(features, centroids, rinv, mc, cc,
                                               out, N, C, D);
}
